// NodeProjection_46677704573242
// MI455X (gfx1250) — matrix operand order repaired and checked by reading, not compiled, not run on hardware
//
#include <hip/hip_runtime.h>

#define DIM    256
#define HID    256
#define NT     4
#define TILE_M 128
// swizzled W fragment array: [NT][kt=8][ntile=16][lane=32][j=8] dwords
#define WB_DWORDS (NT * 8 * 16 * 32 * 8)

typedef __attribute__((ext_vector_type(16))) __bf16 v16bf;
typedef __attribute__((ext_vector_type(8)))  float  v8f;

__device__ __forceinline__ unsigned short f32_to_bf16_rne(float f) {
    unsigned int u = __float_as_uint(f);
    unsigned int r = u + 0x7FFFu + ((u >> 16) & 1u);
    return (unsigned short)(r >> 16);
}
__device__ __forceinline__ float bf16_bits_to_f32(unsigned short h) {
    return __uint_as_float(((unsigned int)h) << 16);
}

// ---------- bucketing kernels ----------
__global__ void zero_k(int* p) {
    if (threadIdx.x < 8) p[threadIdx.x] = 0;   // counts[4] + fill[4]
}

__global__ void hist_k(const int* __restrict__ types, int* __restrict__ counts, int N) {
    int i = blockIdx.x * 256 + threadIdx.x;
    if (i < N) atomicAdd(&counts[types[i] & 3], 1);
}

__global__ void scan_k(const int* __restrict__ counts, int* __restrict__ offsets,
                       int* __restrict__ tstart) {
    if (threadIdx.x == 0 && blockIdx.x == 0) {
        int o = 0, ts = 0;
        for (int t = 0; t < NT; ++t) {
            offsets[t] = o;
            tstart[t]  = ts;
            o  += counts[t];
            ts += (counts[t] + TILE_M - 1) / TILE_M;
        }
        offsets[NT] = o;
        tstart[NT]  = ts;
    }
}

__global__ void scatter_k(const int* __restrict__ types, const int* __restrict__ offsets,
                          int* __restrict__ fill, int* __restrict__ perm, int N) {
    int i = blockIdx.x * 256 + threadIdx.x;
    if (i < N) {
        int t   = types[i] & 3;
        int pos = atomicAdd(&fill[t], 1);
        perm[offsets[t] + pos] = i;
    }
}

// ---------- W -> bf16 hi/lo, pre-swizzled into B-fragment lane layout ----------
// B (32x16 bf16, K x N), this code's k order (same as A): lane n holds column n&15, K-half = (lane<16 ? 0 : 16),
// VGPR j packs K = kbase+2j, kbase+2j+1.  W is torch-style [HID out][DIM in].
__global__ void conv_w_k(const float* __restrict__ W,
                         unsigned int* __restrict__ wbHi,
                         unsigned int* __restrict__ wbLo) {
    int tid = blockIdx.x * 256 + threadIdx.x;
    if (tid >= WB_DWORDS) return;
    int j     = tid & 7;
    int lane  = (tid >> 3)  & 31;
    int ntile = (tid >> 8)  & 15;
    int kt    = (tid >> 12) & 7;
    int t     = (tid >> 15) & 3;
    int n     = ntile * 16 + (lane & 15);
    int kbase = (lane < 16) ? 0 : 16;
    int k0    = kt * 32 + kbase + 2 * j;
    float w0 = W[((size_t)t * HID + n) * DIM + k0];
    float w1 = W[((size_t)t * HID + n) * DIM + k0 + 1];
    unsigned short h0 = f32_to_bf16_rne(w0);
    unsigned short h1 = f32_to_bf16_rne(w1);
    unsigned short l0 = f32_to_bf16_rne(w0 - bf16_bits_to_f32(h0));
    unsigned short l1 = f32_to_bf16_rne(w1 - bf16_bits_to_f32(h1));
    wbHi[tid] = ((unsigned int)h1 << 16) | (unsigned int)h0;
    wbLo[tid] = ((unsigned int)l1 << 16) | (unsigned int)l0;
}

// ---------- main WMMA kernel ----------
// 256 threads = 8 waves. Block tile: 128 gathered rows x 256 cols, single type.
// Wave w owns rows [16w, 16w+16) of the tile, all 16 N-tiles.
__global__ void __launch_bounds__(256)
node_proj_wmma(const float* __restrict__ x,
               const int*   __restrict__ perm,
               const int*   __restrict__ offsets,
               const int*   __restrict__ tstart,
               const unsigned int* __restrict__ wbHi,
               const unsigned int* __restrict__ wbLo,
               const float* __restrict__ bias,
               float*       __restrict__ out) {
    int g = blockIdx.x;
    if (g >= tstart[NT]) return;                 // block-uniform exit
    int t = 0;
    while (t + 1 < NT && g >= tstart[t + 1]) ++t;
    int lt      = g - tstart[t];
    int cnt     = offsets[t + 1] - offsets[t];
    int rowBase = offsets[t] + lt * TILE_M;
    int valid   = cnt - lt * TILE_M;
    if (valid > TILE_M) valid = TILE_M;

    const int lane    = threadIdx.x & 31;
    const int wave    = threadIdx.x >> 5;
    const int halfSel = lane >> 4;       // 0: lanes 0-15, 1: lanes 16-31
    const int lcol    = lane & 15;

    // A-fragment source row for this lane (clamped for ragged last tile)
    int rA     = wave * 16 + lcol;
    int rC     = (rA < valid) ? rA : (valid - 1);
    int nodeA  = perm[rowBase + rC];
    const float4* px = (const float4*)(x + (size_t)nodeA * DIM);

    const unsigned int* wbHt = wbHi + (size_t)t * (8 * 16 * 32 * 8);
    const unsigned int* wbLt = wbLo + (size_t)t * (8 * 16 * 32 * 8);
    const float* bt = bias + t * HID;

    // accumulators init with bias (column-dependent only)
    v8f acc[16];
#pragma unroll
    for (int nt = 0; nt < 16; ++nt) {
        float bv = bt[nt * 16 + lcol];
        v8f a;
#pragma unroll
        for (int e = 0; e < 8; ++e) a[e] = bv;
        acc[nt] = a;
    }

    const int koff = halfSel ? 16 : 0;   // A k order used here (same as B): lanes 0-15 K 0-15; 16-31 K 16-31
    for (int kt = 0; kt < 8; ++kt) {
        int kk = kt * 32;
        float4 f0 = px[(kk + koff) >> 2];
        float4 f1 = px[((kk + koff) >> 2) + 1];
        float4 f2 = px[((kk + koff) >> 2) + 2];
        float4 f3 = px[((kk + koff) >> 2) + 3];
        float vals[16] = {f0.x, f0.y, f0.z, f0.w, f1.x, f1.y, f1.z, f1.w,
                          f2.x, f2.y, f2.z, f2.w, f3.x, f3.y, f3.z, f3.w};
        union { v16bf v; unsigned short s[16]; } ahi, alo;
#pragma unroll
        for (int i = 0; i < 16; ++i) {
            unsigned short h = f32_to_bf16_rne(vals[i]);
            ahi.s[i] = h;
            alo.s[i] = f32_to_bf16_rne(vals[i] - bf16_bits_to_f32(h));
        }
#pragma unroll
        for (int nt = 0; nt < 16; ++nt) {
            size_t bidx = ((size_t)((kt * 16 + nt) * 32 + lane)) * 8;
            union { uint4 q[2]; v16bf v; } bh, bl;
            bh.q[0] = *(const uint4*)(wbHt + bidx);
            bh.q[1] = *(const uint4*)(wbHt + bidx + 4);
            bl.q[0] = *(const uint4*)(wbLt + bidx);
            bl.q[1] = *(const uint4*)(wbLt + bidx + 4);
            // fp32-ish accuracy: hi*hi + lo*hi + hi*lo, fp32 accumulate
            acc[nt] = __builtin_amdgcn_wmma_f32_16x16x32_bf16(
                false, ahi.v, false, bh.v, (short)0, acc[nt], false, false);
            acc[nt] = __builtin_amdgcn_wmma_f32_16x16x32_bf16(
                false, alo.v, false, bh.v, (short)0, acc[nt], false, false);
            acc[nt] = __builtin_amdgcn_wmma_f32_16x16x32_bf16(
                false, ahi.v, false, bl.v, (short)0, acc[nt], false, false);
        }
    }

    // store: D layout — lane col = lcol, VGPR v = row (v + 8*halfSel)
    int  nodeS[8];
    bool okS[8];
#pragma unroll
    for (int v = 0; v < 8; ++v) {
        int M = v + halfSel * 8;
        int r = wave * 16 + M;
        okS[v]   = (r < valid);
        nodeS[v] = perm[rowBase + (okS[v] ? r : 0)];
    }
#pragma unroll
    for (int nt = 0; nt < 16; ++nt) {
#pragma unroll
        for (int v = 0; v < 8; ++v) {
            if (okS[v])
                out[(size_t)nodeS[v] * HID + nt * 16 + lcol] = acc[nt][v];
        }
    }
}

extern "C" void kernel_launch(void* const* d_in, const int* in_sizes, int n_in,
                              void* d_out, int out_size, void* d_ws, size_t ws_size,
                              hipStream_t stream) {
    const float* x     = (const float*)d_in[0];
    const int*   types = (const int*)d_in[1];
    const float* W     = (const float*)d_in[2];
    const float* b     = (const float*)d_in[3];
    float*       out   = (float*)d_out;
    const int N = in_sizes[1];

    char* ws      = (char*)d_ws;
    int*  counts  = (int*)(ws + 0);    // 4 ints
    int*  fill    = (int*)(ws + 16);   // 4 ints
    int*  offsets = (int*)(ws + 32);   // 5 ints
    int*  tstart  = (int*)(ws + 64);   // 5 ints
    int*  perm    = (int*)(ws + 128);  // N ints
    size_t wbOff  = (128 + (size_t)N * 4 + 255) & ~(size_t)255;
    unsigned int* wbHi = (unsigned int*)(ws + wbOff);
    unsigned int* wbLo = wbHi + WB_DWORDS;

    zero_k<<<1, 32, 0, stream>>>(counts);
    hist_k<<<(N + 255) / 256, 256, 0, stream>>>(types, counts, N);
    scan_k<<<1, 1, 0, stream>>>(counts, offsets, tstart);
    scatter_k<<<(N + 255) / 256, 256, 0, stream>>>(types, offsets, fill, perm, N);
    conv_w_k<<<(WB_DWORDS + 255) / 256, 256, 0, stream>>>(W, wbHi, wbLo);

    int tilesMax = (N + TILE_M - 1) / TILE_M + NT;
    node_proj_wmma<<<tilesMax, 256, 0, stream>>>(x, perm, offsets, tstart,
                                                 wbHi, wbLo, b, out);
}
